// GCN_75179107549326
// MI455X (gfx1250) — compile-verified, matrix operand order checked by reading, not run on hardware
//
#include <hip/hip_runtime.h>
#include <hip/hip_bf16.h>

// ---------------- constants ----------------
#define BB   8
#define NN   2048
#define BN   16384        // BB*NN
#define KNB  20           // neighbors
#define CH   64
#define EMB  1024
#define CC   256          // concat channels (4*CH)

typedef __attribute__((ext_vector_type(16))) __bf16   v16bf;
typedef __attribute__((ext_vector_type(8)))  float    v8f;
typedef __attribute__((ext_vector_type(8)))  unsigned v8u;
typedef __attribute__((ext_vector_type(4)))  unsigned v4u;
typedef __attribute__((ext_vector_type(4)))  int      v4i;
typedef __attribute__((address_space(1))) v4i ga_v4i;   // global v4i
typedef __attribute__((address_space(3))) v4i ls_v4i;   // LDS v4i

// Async global->LDS (CDNA5): use if this toolchain exposes the builtins.
#if defined(__has_builtin)
#if __has_builtin(__builtin_amdgcn_global_load_async_to_lds_b128) && \
    __has_builtin(__builtin_amdgcn_s_wait_asynccnt)
#define HAVE_ASYNC_LDS 1
#else
#define HAVE_ASYNC_LDS 0
#endif
#else
#define HAVE_ASYNC_LDS 0
#endif

// ---------------- kernels ----------------

// flat[(b*N+n)*3 + c] = x[b*3*N + c*N + n]
__global__ void k_transpose_flat(const float* __restrict__ x, float* __restrict__ flat) {
  int i = blockIdx.x * blockDim.x + threadIdx.x;
  if (i >= BN * 3) return;
  int row = i / 3, c = i - row * 3;
  int b = row >> 11, n = row & (NN - 1);
  flat[i] = x[b * 3 * NN + c * NN + n];
}

// Faithful quirk: adjacency built from flat buffer re-viewed as (B,3,N):
// pts(b,c,n) = flat_linear[b*3*N + c*N + n]
__global__ void k_knn(const float* __restrict__ flat, int* __restrict__ nbr) {
  int gid = blockIdx.x * blockDim.x + threadIdx.x;
  if (gid >= BN) return;
  int b = gid >> 11, n = gid & (NN - 1);
  const float* base = flat + b * 3 * NN;
  float x0 = base[n], x1 = base[NN + n], x2 = base[2 * NN + n];
  float sc[KNB]; int id[KNB];
#pragma unroll
  for (int j = 0; j < KNB; ++j) { sc[j] = -3.4e38f; id[j] = 0; }
  for (int m = 0; m < NN; ++m) {
    float d0 = x0 - base[m];
    float d1 = x1 - base[NN + m];
    float d2 = x2 - base[2 * NN + m];
    float s = -(d0 * d0 + d1 * d1 + d2 * d2);
    if (s > sc[KNB - 1]) {             // strict > keeps earlier (lower) index on ties
      int j = KNB - 1;
      while (j > 0 && s > sc[j - 1]) { sc[j] = sc[j - 1]; id[j] = id[j - 1]; --j; }
      sc[j] = s; id[j] = m;
    }
  }
#pragma unroll
  for (int j = 0; j < KNB; ++j) nbr[gid * KNB + j] = b * NN + id[j];
}

// out(BN,64) = flat(BN,3) @ w(3,64) [+ bias]
__global__ void k_linear3(const float* __restrict__ flat, const float* __restrict__ w,
                          const float* __restrict__ bias, float* __restrict__ out,
                          int addBias) {
  int i = blockIdx.x * blockDim.x + threadIdx.x;
  if (i >= BN * CH) return;
  int row = i >> 6, c = i & 63;
  float a = flat[row * 3 + 0] * w[c] + flat[row * 3 + 1] * w[CH + c]
          + flat[row * 3 + 2] * w[2 * CH + c];
  out[i] = addBias ? a + bias[c] : a;
}

__global__ void k_fill_bias(float* __restrict__ out, const float* __restrict__ bias) {
  int i = blockIdx.x * blockDim.x + threadIdx.x;
  if (i < BN * CH) out[i] = bias[i & 63];
}

// out[nbr[p,k], :] += tmp[p, :]   (out pre-filled with bias)
__global__ void k_scatter_add(const float* __restrict__ tmp, const int* __restrict__ nbr,
                              float* __restrict__ out) {
  int p = blockIdx.x;
  int c = threadIdx.x;                   // 64 threads
  float v = tmp[p * CH + c];
  const int* nb = nbr + p * KNB;
#pragma unroll
  for (int k = 0; k < KNB; ++k) atomicAdd(&out[nb[k] * CH + c], v);
}

// per-column mean / rstd over `rows` rows (column-major channel reduce)
__global__ void k_bn_stats(const float* __restrict__ x, int rows, int cols,
                           float* __restrict__ mean, float* __restrict__ rstd) {
  int c = blockIdx.x;
  float s = 0.f, s2 = 0.f;
  for (int r = threadIdx.x; r < rows; r += blockDim.x) {
    float v = x[(size_t)r * cols + c];
    s += v; s2 += v * v;
  }
  __shared__ float sh0[256], sh1[256];
  sh0[threadIdx.x] = s; sh1[threadIdx.x] = s2;
  __syncthreads();
  for (int off = 128; off > 0; off >>= 1) {
    if ((int)threadIdx.x < off) {
      sh0[threadIdx.x] += sh0[threadIdx.x + off];
      sh1[threadIdx.x] += sh1[threadIdx.x + off];
    }
    __syncthreads();
  }
  if (threadIdx.x == 0) {
    float m = sh0[0] / rows;
    float var = sh1[0] / rows - m * m;
    mean[c] = m;
    rstd[c] = rsqrtf(var + 1e-5f);
  }
}

// h = relu( relu(bn(o_pre)) + bn(res_pre) )
__global__ void k_combine0(const float* __restrict__ opre, const float* __restrict__ mA,
                           const float* __restrict__ rA, const float* __restrict__ g0,
                           const float* __restrict__ be0,
                           const float* __restrict__ rpre, const float* __restrict__ mB,
                           const float* __restrict__ rB, const float* __restrict__ gr,
                           const float* __restrict__ ber, float* __restrict__ h) {
  int i = blockIdx.x * blockDim.x + threadIdx.x;
  if (i >= BN * CH) return;
  int c = i & 63;
  float o = (opre[i] - mA[c]) * rA[c] * g0[c] + be0[c];
  o = fmaxf(o, 0.f);
  float r = (rpre[i] - mB[c]) * rB[c] * gr[c] + ber[c];
  h[i] = fmaxf(o + r, 0.f);
}

// h_new = relu( relu(bn(o_pre)) + h_old )
__global__ void k_combineN(const float* __restrict__ opre, const float* __restrict__ m,
                           const float* __restrict__ r, const float* __restrict__ g,
                           const float* __restrict__ be, const float* __restrict__ hold,
                           float* __restrict__ hnew) {
  int i = blockIdx.x * blockDim.x + threadIdx.x;
  if (i >= BN * CH) return;
  int c = i & 63;
  float o = (opre[i] - m[c]) * r[c] * g[c] + be[c];
  o = fmaxf(o, 0.f);
  hnew[i] = fmaxf(o + hold[i], 0.f);
}

__global__ void k_cast_bf16(const float* __restrict__ in, __bf16* __restrict__ out, int n) {
  int i = blockIdx.x * blockDim.x + threadIdx.x;
  if (i < n) out[i] = (__bf16)in[i];
}

// wt[n*Kd + k] = (bf16) w[k*Nc + n]
__global__ void k_transpose_cast_w(const float* __restrict__ w, __bf16* __restrict__ wt,
                                   int Kd, int Nc) {
  int i = blockIdx.x * blockDim.x + threadIdx.x;
  if (i >= Kd * Nc) return;
  int n = i / Kd, k = i - n * Kd;
  wt[i] = (__bf16)w[k * Nc + n];
}

// F[row, c] = outs[c/64][row, c%64]   (bf16)
__global__ void k_build_F(const float* __restrict__ o0, const float* __restrict__ o1,
                          const float* __restrict__ o2, const float* __restrict__ o3,
                          __bf16* __restrict__ F) {
  int i = blockIdx.x * blockDim.x + threadIdx.x;
  if (i >= BN * CC) return;
  int row = i >> 8, c = i & 255;
  const float* src = (c < 64) ? o0 : (c < 128) ? o1 : (c < 192) ? o2 : o3;
  F[i] = (__bf16)src[row * CH + (c & 63)];
}

// Simple GEMM for small-N layers: one wave per 16x16 tile.
__global__ void k_wmma_gemm_bf16(const __bf16* __restrict__ A,
                                 const __bf16* __restrict__ Bt,
                                 float* __restrict__ C,
                                 int M, int Kd, int Nc) {
  int wavesPerBlock = blockDim.x >> 5;
  int wave = blockIdx.x * wavesPerBlock + ((int)threadIdx.x >> 5);
  int tilesN = Nc >> 4;
  int tM = (wave / tilesN) << 4;
  int tN = (wave - (wave / tilesN) * tilesN) << 4;
  if (tM >= M) return;
  int lane = threadIdx.x & 31;
  int half = lane >> 4;
  int mr   = lane & 15;
  v8f acc = {};
  const __bf16* arow = A  + (size_t)(tM + mr) * Kd;
  const __bf16* brow = Bt + (size_t)(tN + mr) * Kd;
  for (int k0 = 0; k0 < Kd; k0 += 32) {
    v8u av, bv;
#pragma unroll
    for (int v = 0; v < 8; ++v) {
      // element pair (2v,2v+1) of this lane holds k = (2v&7) + 8*half + 16*(v>>2)
      int kk = k0 + ((2 * v) & 7) + 8 * half + 16 * (v >> 2);
      av[v] = *(const unsigned*)(arow + kk);
      bv[v] = *(const unsigned*)(brow + kk);
    }
    v16bf a = __builtin_bit_cast(v16bf, av);
    v16bf b = __builtin_bit_cast(v16bf, bv);
    acc = __builtin_amdgcn_wmma_f32_16x16x32_bf16(false, a, false, b,
                                                  (short)0, acc, false, false);
  }
#pragma unroll
  for (int r = 0; r < 8; ++r)
    C[(size_t)(tM + r + 8 * half) * Nc + tN + mr] = acc[r];
}

// Big GEMM: 128x128 C-tile per block (8 waves). Wave w owns rows [w*16, w*16+16)
// x 128 cols -> 8 f32 accumulators; A fragment loaded once per k-step and reused
// across 8 back-to-back WMMAs. B fragments for the whole block staged in LDS
// (async global->LDS when available), shared by all 8 waves.
__global__ __launch_bounds__(256) void k_wmma_gemm_tile128(
    const __bf16* __restrict__ A, const __bf16* __restrict__ Bt,
    float* __restrict__ C, int M, int Kd, int Nc) {
  __shared__ v4u smem4[8 * 32 * 2];      // 8 subtiles x 32 lanes x 32B = 8 KB
  int tilesN = Nc >> 7;
  int bm = blockIdx.x / tilesN;
  int bn = blockIdx.x - bm * tilesN;
  int tM0 = bm << 7, tN0 = bn << 7;
  int tid  = threadIdx.x;
  int wv   = tid >> 5;                   // wave id == staged subtile id
  int lane = tid & 31;
  int half = lane >> 4, mr = lane & 15;

  v8f acc[8];
#pragma unroll
  for (int t = 0; t < 8; ++t) acc[t] = (v8f){};

  const __bf16* arow = A  + (size_t)(tM0 + (wv << 4) + mr) * Kd + 8 * half;
  const __bf16* brow = Bt + (size_t)(tN0 + (wv << 4) + mr) * Kd + 8 * half;

  for (int k0 = 0; k0 < Kd; k0 += 32) {
    // ---- stage B fragments (this thread: lane `lane` of subtile `wv`) ----
    {
      const __bf16* src = brow + k0;
      v4u* dst = &smem4[(wv * 32 + lane) * 2];
#if HAVE_ASYNC_LDS
      __builtin_amdgcn_global_load_async_to_lds_b128(
          (ga_v4i*)src, (ls_v4i*)dst, 0, 0);
      __builtin_amdgcn_global_load_async_to_lds_b128(
          (ga_v4i*)(src + 16), (ls_v4i*)(dst + 1), 0, 0);
      __builtin_amdgcn_s_wait_asynccnt(0);
#else
      dst[0] = *(const v4u*)src;
      dst[1] = *(const v4u*)(src + 16);
#endif
    }
    __syncthreads();

    // ---- A fragment: two contiguous 16B chunks ----
    const __bf16* ap = arow + k0;
    if (k0 + 32 < Kd) __builtin_prefetch((const void*)(ap + 32), 0, 1);
    v4u alo = *(const v4u*)ap;
    v4u ahi = *(const v4u*)(ap + 16);
    v16bf a = __builtin_bit_cast(
        v16bf, __builtin_shufflevector(alo, ahi, 0, 1, 2, 3, 4, 5, 6, 7));

    // ---- 8 WMMAs reusing the A fragment ----
#pragma unroll
    for (int t = 0; t < 8; ++t) {
      const v4u* fb = &smem4[(t * 32 + lane) * 2];
      v4u blo = fb[0], bhi = fb[1];
      v16bf b = __builtin_bit_cast(
          v16bf, __builtin_shufflevector(blo, bhi, 0, 1, 2, 3, 4, 5, 6, 7));
      acc[t] = __builtin_amdgcn_wmma_f32_16x16x32_bf16(false, a, false, b,
                                                       (short)0, acc[t], false, false);
    }
    __syncthreads();
  }

  // ---- store 16x128 per wave ----
#pragma unroll
  for (int t = 0; t < 8; ++t) {
    float* cp = C + (size_t)(tM0 + (wv << 4)) * Nc + tN0 + (t << 4) + mr;
#pragma unroll
    for (int r = 0; r < 8; ++r)
      cp[(size_t)(r + 8 * half) * Nc] = acc[t][r];
  }
}

// bn + leaky(0.2) + per-(b,e) max & mean over n -> h2(8, 2048)
__global__ void k_pool(const float* __restrict__ Y, const float* __restrict__ mean,
                       const float* __restrict__ rstd, const float* __restrict__ gf,
                       const float* __restrict__ bef, float* __restrict__ h2) {
  int b = blockIdx.x >> 10;
  int e = blockIdx.x & (EMB - 1);
  float m = mean[e], r = rstd[e], g = gf[e], be = bef[e];
  float mx = -3.4e38f, sm = 0.f;
  for (int n = threadIdx.x; n < NN; n += blockDim.x) {
    float v = Y[(size_t)(b * NN + n) * EMB + e];
    v = (v - m) * r * g + be;
    v = v > 0.f ? v : 0.2f * v;
    mx = fmaxf(mx, v);
    sm += v;
  }
  __shared__ float shm[256], shs[256];
  shm[threadIdx.x] = mx; shs[threadIdx.x] = sm;
  __syncthreads();
  for (int off = 128; off > 0; off >>= 1) {
    if ((int)threadIdx.x < off) {
      shm[threadIdx.x] = fmaxf(shm[threadIdx.x], shm[threadIdx.x + off]);
      shs[threadIdx.x] += shs[threadIdx.x + off];
    }
    __syncthreads();
  }
  if (threadIdx.x == 0) {
    h2[b * 2 * EMB + e]       = shm[0];
    h2[b * 2 * EMB + EMB + e] = shs[0] / (float)NN;
  }
}

// out(8,O) = in(8,Kd) @ w(Kd,O) [+ bias]
__global__ void k_head_gemm(const float* __restrict__ in, const float* __restrict__ w,
                            const float* __restrict__ bias, float* __restrict__ out,
                            int Kd, int O, int addBias) {
  int i = blockIdx.x * blockDim.x + threadIdx.x;
  if (i >= 8 * O) return;
  int b = i / O, o = i - b * O;
  float acc = addBias ? bias[o] : 0.f;
  for (int k = 0; k < Kd; ++k) acc += in[b * Kd + k] * w[k * O + o];
  out[i] = acc;
}

// per-column BN over 8 rows + leaky(0.2)
__global__ void k_head_bn_leaky(const float* __restrict__ in, const float* __restrict__ g,
                                const float* __restrict__ be, float* __restrict__ out,
                                int cols) {
  int c = blockIdx.x * blockDim.x + threadIdx.x;
  if (c >= cols) return;
  float s = 0.f, s2 = 0.f;
#pragma unroll
  for (int b = 0; b < 8; ++b) { float v = in[b * cols + c]; s += v; s2 += v * v; }
  float m = s * 0.125f;
  float var = s2 * 0.125f - m * m;
  float r = rsqrtf(var + 1e-5f);
#pragma unroll
  for (int b = 0; b < 8; ++b) {
    float v = (in[b * cols + c] - m) * r * g[c] + be[c];
    out[b * cols + c] = v > 0.f ? v : 0.2f * v;
  }
}

// ---------------- launch ----------------
extern "C" void kernel_launch(void* const* d_in, const int* in_sizes, int n_in,
                              void* d_out, int out_size, void* d_ws, size_t ws_size,
                              hipStream_t stream) {
  const float* x     = (const float*)d_in[0];
  const float* w0    = (const float*)d_in[1];
  const float* b0    = (const float*)d_in[2];
  const float* g0    = (const float*)d_in[3];
  const float* be0   = (const float*)d_in[4];
  const float* wr    = (const float*)d_in[5];
  const float* br    = (const float*)d_in[6];
  const float* gr    = (const float*)d_in[7];
  const float* ber   = (const float*)d_in[8];
  const float* W     = (const float*)d_in[9];    // (3,64,64)
  const float* Bv    = (const float*)d_in[10];   // (3,64)
  const float* G     = (const float*)d_in[11];
  const float* Be    = (const float*)d_in[12];
  const float* wconv = (const float*)d_in[13];   // (1024,256)
  const float* gf    = (const float*)d_in[14];
  const float* bef   = (const float*)d_in[15];
  const float* w1    = (const float*)d_in[16];   // (2048,512)
  const float* g6    = (const float*)d_in[17];
  const float* b6    = (const float*)d_in[18];
  const float* w2    = (const float*)d_in[19];   // (512,256)
  const float* b2    = (const float*)d_in[20];
  const float* g7    = (const float*)d_in[21];
  const float* b7    = (const float*)d_in[22];
  const float* w3    = (const float*)d_in[23];   // (256,40)
  const float* b3    = (const float*)d_in[24];
  float* out = (float*)d_out;

  char* ws = (char*)d_ws;
  size_t off = 0;
  auto alloc = [&](size_t bytes) -> void* {
    off = (off + 255) & ~(size_t)255;
    void* p = ws + off;
    off += bytes;
    return p;
  };

  float*   FLAT = (float*)alloc((size_t)BN * 3 * 4);
  int*     NBR  = (int*)  alloc((size_t)BN * KNB * 4);
  float*   H[4];
  for (int i = 0; i < 4; ++i) H[i] = (float*)alloc((size_t)BN * CH * 4);
  float*   TMP  = (float*)alloc((size_t)BN * CH * 4);   // gemm output
  float*   OPRE = (float*)alloc((size_t)BN * CH * 4);   // scatter target
  float*   RES  = (float*)alloc((size_t)BN * CH * 4);   // layer0 residual
  __bf16*  A16  = (__bf16*)alloc((size_t)BN * CH * 2);
  __bf16*  WT16 = (__bf16*)alloc((size_t)CC * EMB * 2); // max weight: wconv
  __bf16*  F16  = (__bf16*)alloc((size_t)BN * CC * 2);
  float*   Y    = (float*)alloc((size_t)BN * EMB * 4);  // 64 MB
  float*   MEAN = (float*)alloc(EMB * 4);
  float*   RSTD = (float*)alloc(EMB * 4);
  float*   MEAN2= (float*)alloc(EMB * 4);
  float*   RSTD2= (float*)alloc(EMB * 4);
  float*   H2   = (float*)alloc((size_t)8 * 2 * EMB * 4);
  float*   T1P  = (float*)alloc((size_t)8 * 512 * 4);
  float*   T1   = (float*)alloc((size_t)8 * 512 * 4);
  float*   T2P  = (float*)alloc((size_t)8 * 256 * 4);
  float*   T2   = (float*)alloc((size_t)8 * 256 * 4);
  (void)ws_size; (void)n_in; (void)in_sizes; (void)out_size;

  const int TB = 256;
  // ---- preprocessing ----
  k_transpose_flat<<<(BN * 3 + TB - 1) / TB, TB, 0, stream>>>(x, FLAT);
  k_knn<<<(BN + 127) / 128, 128, 0, stream>>>(FLAT, NBR);

  // ---- layer 0 (in=3) with projection residual ----
  int elemsHC = BN * CH;
  k_linear3<<<(elemsHC + TB - 1) / TB, TB, 0, stream>>>(FLAT, w0, b0, TMP, 0);
  k_fill_bias<<<(elemsHC + TB - 1) / TB, TB, 0, stream>>>(OPRE, b0);
  k_scatter_add<<<BN, CH, 0, stream>>>(TMP, NBR, OPRE);
  k_linear3<<<(elemsHC + TB - 1) / TB, TB, 0, stream>>>(FLAT, wr, br, RES, 1);
  k_bn_stats<<<CH, TB, 0, stream>>>(OPRE, BN, CH, MEAN, RSTD);
  k_bn_stats<<<CH, TB, 0, stream>>>(RES,  BN, CH, MEAN2, RSTD2);
  k_combine0<<<(elemsHC + TB - 1) / TB, TB, 0, stream>>>(
      OPRE, MEAN, RSTD, g0, be0, RES, MEAN2, RSTD2, gr, ber, H[0]);

  // ---- layers 1..3 (64->64, WMMA bf16) ----
  for (int i = 0; i < 3; ++i) {
    k_cast_bf16<<<(elemsHC + TB - 1) / TB, TB, 0, stream>>>(H[i], A16, elemsHC);
    k_transpose_cast_w<<<(CH * CH + TB - 1) / TB, TB, 0, stream>>>(
        W + (size_t)i * CH * CH, WT16, CH, CH);
    int waves = (BN / 16) * (CH / 16);      // 4096
    k_wmma_gemm_bf16<<<waves / 8, TB, 0, stream>>>(A16, WT16, TMP, BN, CH, CH);
    k_fill_bias<<<(elemsHC + TB - 1) / TB, TB, 0, stream>>>(OPRE, Bv + i * CH);
    k_scatter_add<<<BN, CH, 0, stream>>>(TMP, NBR, OPRE);
    k_bn_stats<<<CH, TB, 0, stream>>>(OPRE, BN, CH, MEAN, RSTD);
    k_combineN<<<(elemsHC + TB - 1) / TB, TB, 0, stream>>>(
        OPRE, MEAN, RSTD, G + i * CH, Be + i * CH, H[i], H[i + 1]);
  }

  // ---- conv_final: Y(16384,1024) = F(16384,256) @ wconv^T ----
  k_build_F<<<(BN * CC + TB - 1) / TB, TB, 0, stream>>>(H[0], H[1], H[2], H[3], F16);
  k_cast_bf16<<<(EMB * CC + TB - 1) / TB, TB, 0, stream>>>(wconv, WT16, EMB * CC);
  k_wmma_gemm_tile128<<<(BN / 128) * (EMB / 128), 256, 0, stream>>>(
      F16, WT16, Y, BN, CC, EMB);
  k_bn_stats<<<EMB, TB, 0, stream>>>(Y, BN, EMB, MEAN, RSTD);
  k_pool<<<BB * EMB, TB, 0, stream>>>(Y, MEAN, RSTD, gf, bef, H2);

  // ---- head ----
  k_head_gemm<<<(8 * 512 + TB - 1) / TB, TB, 0, stream>>>(H2, w1, nullptr, T1P, 2 * EMB, 512, 0);
  k_head_bn_leaky<<<(512 + TB - 1) / TB, TB, 0, stream>>>(T1P, g6, b6, T1, 512);
  k_head_gemm<<<(8 * 256 + TB - 1) / TB, TB, 0, stream>>>(T1, w2, b2, T2P, 512, 256, 1);
  k_head_bn_leaky<<<(256 + TB - 1) / TB, TB, 0, stream>>>(T2P, g7, b7, T2, 256);
  k_head_gemm<<<(8 * 40 + TB - 1) / TB, TB, 0, stream>>>(T2, w3, b3, out, 256, 40, 1);
}
